// GPT_12738873000105
// MI455X (gfx1250) — matrix operand order repaired and checked by reading, not compiled, not run on hardware
//
#include <hip/hip_runtime.h>
#include <hip/hip_bf16.h>

typedef __attribute__((ext_vector_type(16))) __bf16 v16bf;
typedef __attribute__((ext_vector_type(8)))  float  v8f;

union Frag16 {
  v16bf v;
  uint4 q[2];
};

__device__ __forceinline__ unsigned short f32_to_bf16_rne(float f) {
  unsigned int u = __builtin_bit_cast(unsigned int, f);
  unsigned int r = u + 0x7FFFu + ((u >> 16) & 1u);
  return (unsigned short)(r >> 16);
}

__device__ __forceinline__ unsigned int pack_bf16x2(float lo, float hi) {
  return (unsigned int)f32_to_bf16_rne(lo) | ((unsigned int)f32_to_bf16_rne(hi) << 16);
}

__device__ __forceinline__ v8f wmma_bf16(const Frag16& a, const Frag16& b, v8f c) {
  return __builtin_amdgcn_wmma_f32_16x16x32_bf16(false, a.v, false, b.v,
                                                 (short)0, c, false, false);
}

// -------------------- embedding: x = tok_emb[idx] + pos_emb --------------------
__global__ void k_embed(const int* __restrict__ idx, const float* __restrict__ tok,
                        const float* __restrict__ pos, float* __restrict__ x) {
  int m = blockIdx.x;            // 0..2047, m = b*1024 + t
  int t = m & 1023;
  int id = idx[m];
  for (int e = threadIdx.x; e < 768; e += 256)
    x[(long long)m * 768 + e] = tok[(long long)id * 768 + e] + pos[(long long)t * 768 + e];
}

// -------------------- layernorm -> bf16 --------------------
__global__ void k_layernorm(const float* __restrict__ x, const float* __restrict__ g,
                            const float* __restrict__ bta, unsigned short* __restrict__ out) {
  __shared__ float red[256];
  int m = blockIdx.x;
  const float* xr = x + (long long)m * 768;
  int tid = threadIdx.x;
  float v0 = xr[tid], v1 = xr[tid + 256], v2 = xr[tid + 512];
  red[tid] = v0 + v1 + v2;
  __syncthreads();
  for (int o = 128; o > 0; o >>= 1) {
    if (tid < o) red[tid] += red[tid + o];
    __syncthreads();
  }
  float mean = red[0] * (1.0f / 768.0f);
  __syncthreads();
  float d0 = v0 - mean, d1 = v1 - mean, d2 = v2 - mean;
  red[tid] = d0 * d0 + d1 * d1 + d2 * d2;
  __syncthreads();
  for (int o = 128; o > 0; o >>= 1) {
    if (tid < o) red[tid] += red[tid + o];
    __syncthreads();
  }
  float rstd = rsqrtf(red[0] * (1.0f / 768.0f) + 1e-5f);
  unsigned short* outr = out + (long long)m * 768;
  outr[tid]       = f32_to_bf16_rne(d0 * rstd * g[tid]       + bta[tid]);
  outr[tid + 256] = f32_to_bf16_rne(d1 * rstd * g[tid + 256] + bta[tid + 256]);
  outr[tid + 512] = f32_to_bf16_rne(d2 * rstd * g[tid + 512] + bta[tid + 512]);
}

// wave computes 32x32: 8 WMMAs per 64-K chunk, 2 ds_loads per WMMA
__device__ __forceinline__ void mma_k64_2x2(const unsigned short (*As)[72],
                                            const unsigned short (*Ws)[72],
                                            int mw, int nw, int r, int aoff, int klo,
                                            v8f& acc00, v8f& acc01,
                                            v8f& acc10, v8f& acc11) {
#pragma unroll
  for (int kx = 0; kx < 2; ++kx) {
    const int kk = kx * 32;
    Frag16 a0, a1, b0, b1;
    a0.q[0] = *(const uint4*)&As[mw + r][kk + aoff];
    a0.q[1] = *(const uint4*)&As[mw + r][kk + 16 + aoff];
    a1.q[0] = *(const uint4*)&As[mw + 16 + r][kk + aoff];
    a1.q[1] = *(const uint4*)&As[mw + 16 + r][kk + 16 + aoff];
    b0.q[0] = *(const uint4*)&Ws[nw + r][kk + aoff];
    b0.q[1] = *(const uint4*)&Ws[nw + r][kk + 16 + aoff];
    b1.q[0] = *(const uint4*)&Ws[nw + 16 + r][kk + aoff];
    b1.q[1] = *(const uint4*)&Ws[nw + 16 + r][kk + 16 + aoff];
    acc00 = wmma_bf16(a0, b0, acc00);
    acc01 = wmma_bf16(a0, b1, acc01);
    acc10 = wmma_bf16(a1, b0, acc10);
    acc11 = wmma_bf16(a1, b1, acc11);
  }
}

// -------------------- generic bf16 WMMA GEMM --------------------
// C[M,N] = A_bf16[M,K] @ W_f32->bf16[K,N] (+bias) (+resid) (relu?) -> f32 and/or bf16
// 256 threads, 128x64 block tile, 8 waves as 4(m)x2(n), each wave 32x32 output.
// KC=64 (K must be a multiple of 64). Software-pipelined global->LDS staging.
__global__ __launch_bounds__(256)
void k_gemm(const unsigned short* __restrict__ A,
            const float* __restrict__ W,
            const float* __restrict__ bias,
            const float* __restrict__ resid,
            float* __restrict__ outF,
            unsigned short* __restrict__ outB,
            int N, int K, int relu,
            long long wStride, long long outStride) {
  __shared__ unsigned short As[128][72];  // [m][k] bf16, 144B row (16B aligned)
  __shared__ unsigned short Ws[64][72];   // [n][k] bf16 (transposed)
  const int z = blockIdx.z;
  W += (long long)z * wStride;
  const int tid = threadIdx.x;
  const int lane = tid & 31;
  const int wv = tid >> 5;          // 0..7
  const int m0 = blockIdx.y * 128;
  const int n0 = blockIdx.x * 64;
  const int atm = tid >> 1;         // A staging row 0..127
  const int atc = (tid & 1) * 32;   // A staging col 0 or 32
  const int wk  = (tid >> 4) * 4;   // W staging k 0..60
  const int wng = (tid & 15) * 4;   // W staging n 0..60
  const int r = lane & 15;
  const int half = lane >> 4;
  const int mw = (wv >> 1) * 32;    // wave M base (0,32,64,96)
  const int nw = (wv & 1) * 32;     // wave N base (0 or 32)
  const int aoff = half * 8;        // A frag K offset per ISA layout
  const int klo = half * 16;        // unused: B frags use aoff (same K order as A)
  const bool vecN = ((N & 3) == 0) && (n0 + 64 <= N);
  v8f acc00 = {0.f, 0.f, 0.f, 0.f, 0.f, 0.f, 0.f, 0.f};
  v8f acc01 = acc00, acc10 = acc00, acc11 = acc00;

  const unsigned short* aptr = &A[(long long)(m0 + atm) * K + atc];

  if (vecN) {
    const float* wbase = W + n0 + wng;
    uint4 ra0 = *(const uint4*)aptr;
    uint4 ra1 = *(const uint4*)(aptr + 8);
    uint4 ra2 = *(const uint4*)(aptr + 16);
    uint4 ra3 = *(const uint4*)(aptr + 24);
    const float* wp = wbase + (long long)wk * N;
    float4 f0 = *(const float4*)wp;
    float4 f1 = *(const float4*)(wp + N);
    float4 f2 = *(const float4*)(wp + 2 * N);
    float4 f3 = *(const float4*)(wp + 3 * N);
    for (int k0 = 0; k0 < K; k0 += 64) {
      __syncthreads();
      *(uint4*)&As[atm][atc] = ra0;
      *(uint4*)&As[atm][atc + 8] = ra1;
      *(uint4*)&As[atm][atc + 16] = ra2;
      *(uint4*)&As[atm][atc + 24] = ra3;
      const float* p0 = (const float*)&f0;
      const float* p1 = (const float*)&f1;
      const float* p2 = (const float*)&f2;
      const float* p3 = (const float*)&f3;
#pragma unroll
      for (int j = 0; j < 4; ++j) {
        *(unsigned int*)&Ws[wng + j][wk]     = pack_bf16x2(p0[j], p1[j]);
        *(unsigned int*)&Ws[wng + j][wk + 2] = pack_bf16x2(p2[j], p3[j]);
      }
      const int kn = (k0 + 64 < K) ? (k0 + 64) : k0;   // clamped prefetch (uniform)
      const unsigned short* ap = aptr + kn;
      ra0 = *(const uint4*)ap;
      ra1 = *(const uint4*)(ap + 8);
      ra2 = *(const uint4*)(ap + 16);
      ra3 = *(const uint4*)(ap + 24);
      wp = wbase + (long long)(kn + wk) * N;
      f0 = *(const float4*)wp;
      f1 = *(const float4*)(wp + N);
      f2 = *(const float4*)(wp + 2 * N);
      f3 = *(const float4*)(wp + 3 * N);
      __syncthreads();
      mma_k64_2x2(As, Ws, mw, nw, r, aoff, klo, acc00, acc01, acc10, acc11);
    }
  } else {
    // branch-free ragged-N path: clamp column index, mask value (no exec divergence)
    int ncol[4];
    float nmask[4];
#pragma unroll
    for (int j = 0; j < 4; ++j) {
      int n = n0 + wng + j;
      nmask[j] = (n < N) ? 1.0f : 0.0f;
      ncol[j] = (n < N) ? n : (N - 1);
    }
    uint4 ra0 = *(const uint4*)aptr;
    uint4 ra1 = *(const uint4*)(aptr + 8);
    uint4 ra2 = *(const uint4*)(aptr + 16);
    uint4 ra3 = *(const uint4*)(aptr + 24);
    float wreg[16];
    {
      const float* wr = W + (long long)wk * N;
#pragma unroll
      for (int j = 0; j < 4; ++j) {
        wreg[4 * j + 0] = wr[ncol[j]];
        wreg[4 * j + 1] = wr[N + ncol[j]];
        wreg[4 * j + 2] = wr[2 * N + ncol[j]];
        wreg[4 * j + 3] = wr[3 * N + ncol[j]];
      }
    }
    for (int k0 = 0; k0 < K; k0 += 64) {
      __syncthreads();
      *(uint4*)&As[atm][atc] = ra0;
      *(uint4*)&As[atm][atc + 8] = ra1;
      *(uint4*)&As[atm][atc + 16] = ra2;
      *(uint4*)&As[atm][atc + 24] = ra3;
#pragma unroll
      for (int j = 0; j < 4; ++j) {
        *(unsigned int*)&Ws[wng + j][wk] =
            pack_bf16x2(wreg[4 * j + 0] * nmask[j], wreg[4 * j + 1] * nmask[j]);
        *(unsigned int*)&Ws[wng + j][wk + 2] =
            pack_bf16x2(wreg[4 * j + 2] * nmask[j], wreg[4 * j + 3] * nmask[j]);
      }
      const int kn = (k0 + 64 < K) ? (k0 + 64) : k0;
      const unsigned short* ap = aptr + kn;
      ra0 = *(const uint4*)ap;
      ra1 = *(const uint4*)(ap + 8);
      ra2 = *(const uint4*)(ap + 16);
      ra3 = *(const uint4*)(ap + 24);
      const float* wr = W + (long long)(kn + wk) * N;
#pragma unroll
      for (int j = 0; j < 4; ++j) {
        wreg[4 * j + 0] = wr[ncol[j]];
        wreg[4 * j + 1] = wr[N + ncol[j]];
        wreg[4 * j + 2] = wr[2 * N + ncol[j]];
        wreg[4 * j + 3] = wr[3 * N + ncol[j]];
      }
      __syncthreads();
      mma_k64_2x2(As, Ws, mw, nw, r, aoff, klo, acc00, acc01, acc10, acc11);
    }
  }

#pragma unroll
  for (int mi = 0; mi < 2; ++mi) {
#pragma unroll
    for (int ni = 0; ni < 2; ++ni) {
      const v8f& acc = mi ? (ni ? acc11 : acc10) : (ni ? acc01 : acc00);
      const int col = n0 + nw + ni * 16 + r;
      if (col < N) {
        float bv = bias ? bias[col] : 0.0f;
#pragma unroll
        for (int i = 0; i < 8; ++i) {
          int row = m0 + mw + mi * 16 + half * 8 + i;
          float c = acc[i] + bv;
          if (resid) c += resid[(long long)row * N + col];
          if (relu) c = fmaxf(c, 0.0f);
          long long oi = (long long)z * outStride + (long long)row * N + col;
          if (outF) outF[oi] = c;
          if (outB) outB[oi] = f32_to_bf16_rne(c);
        }
      }
    }
  }
}

// -------------------- flash attention --------------------
// scores[t,s] = k[t] . q[s] * SCALE, causal s<=t, o = softmax(scores) @ v
// q/k/v layout: [h][b*T+t][64] bf16.  grid: (T/64, B*H), 128 threads (4 waves x 16 rows)
__global__ __launch_bounds__(128)
void k_attn(const unsigned short* __restrict__ q,
            const unsigned short* __restrict__ k,
            const unsigned short* __restrict__ v,
            unsigned short* __restrict__ o) {
  __shared__ unsigned short VT[64][40];     // [d][s] transposed V chunk
  __shared__ unsigned short P[4][16][40];   // per-wave prob tile [t][s]
  const int bh = blockIdx.y;
  const int b = bh / 12, h = bh % 12;
  const long long base = (long long)h * (2048 * 64) + (long long)b * (1024 * 64);
  const int tid = threadIdx.x;
  const int lane = tid & 31;
  const int wv = tid >> 5;
  const int tb = blockIdx.x * 64;
  const int tw0 = tb + wv * 16;
  const int r = lane & 15;
  const int half = lane >> 4;
  const int aoff = half * 8;
  const int klo = half * 16;
  const float scale = 1.0f / (768.0f * 768.0f);
  const int vsp = tid >> 3;        // V staging: s-pair 0..15
  const int vd0 = (tid & 7) * 8;   // V staging: d 0..56

  // K-rows are fixed for this wave: hoist the A fragments (d=0..31 and d=32..63)
  Frag16 a0, a1;
  {
    const unsigned short* kp = &k[base + (long long)(tw0 + r) * 64];
    a0.q[0] = *(const uint4*)(kp + aoff);
    a0.q[1] = *(const uint4*)(kp + 16 + aoff);
    a1.q[0] = *(const uint4*)(kp + 32 + aoff);
    a1.q[1] = *(const uint4*)(kp + 48 + aoff);
  }

  v8f O0 = {0.f,0.f,0.f,0.f,0.f,0.f,0.f,0.f};
  v8f O1 = O0, O2 = O0, O3 = O0;
  float mrun[8], lrun[8];
#pragma unroll
  for (int i = 0; i < 8; ++i) { mrun[i] = -3.0e38f; lrun[i] = 0.0f; }

  // prefetch V chunk 0
  const unsigned short* vstage = &v[base + (long long)(2 * vsp) * 64 + vd0];
  uint4 rv0 = *(const uint4*)vstage;
  uint4 rv1 = *(const uint4*)(vstage + 64);

  const int send = tb + 64;
  for (int sc = 0; sc < send; sc += 32) {
    __syncthreads();
    // stage V chunk transposed: VT[d][s-sc], bf16 bits repacked in s-pairs
    {
      const unsigned short* h0 = (const unsigned short*)&rv0;
      const unsigned short* h1 = (const unsigned short*)&rv1;
#pragma unroll
      for (int j = 0; j < 8; ++j)
        *(unsigned int*)&VT[vd0 + j][2 * vsp] =
            (unsigned int)h0[j] | ((unsigned int)h1[j] << 16);
    }
    // prefetch next V chunk (clamped, uniform)
    {
      const int scn = (sc + 32 < send) ? (sc + 32) : sc;
      const unsigned short* vp = vstage + (long long)scn * 64;
      rv0 = *(const uint4*)vp;
      rv1 = *(const uint4*)(vp + 64);
    }
    __syncthreads();

    // S tiles: 16(t) x 32(s), all 4 q B-frags loaded, then 4 WMMAs
    v8f S0 = {0.f,0.f,0.f,0.f,0.f,0.f,0.f,0.f};
    v8f S1 = S0;
    {
      Frag16 bq00, bq01, bq10, bq11;
      const unsigned short* qp0 = &q[base + (long long)(sc + r) * 64];
      const unsigned short* qp1 = qp0 + 16 * 64;
      bq00.q[0] = *(const uint4*)(qp0 + aoff);
      bq00.q[1] = *(const uint4*)(qp0 + 16 + aoff);
      bq01.q[0] = *(const uint4*)(qp0 + 32 + aoff);
      bq01.q[1] = *(const uint4*)(qp0 + 48 + aoff);
      bq10.q[0] = *(const uint4*)(qp1 + aoff);
      bq10.q[1] = *(const uint4*)(qp1 + 16 + aoff);
      bq11.q[0] = *(const uint4*)(qp1 + 32 + aoff);
      bq11.q[1] = *(const uint4*)(qp1 + 48 + aoff);
      S0 = wmma_bf16(a0, bq00, S0);
      S1 = wmma_bf16(a0, bq10, S1);
      S0 = wmma_bf16(a1, bq01, S0);
      S1 = wmma_bf16(a1, bq11, S1);
    }

    // scale + causal mask + online softmax (row stats via 16-lane shfl reductions)
    const int col0 = sc + r;
    const int col1 = sc + 16 + r;
#pragma unroll
    for (int i = 0; i < 8; ++i) {
      int trow = tw0 + half * 8 + i;
      float s0 = (col0 <= trow) ? S0[i] * scale : -3.0e38f;
      float s1 = (col1 <= trow) ? S1[i] * scale : -3.0e38f;
      float mx = fmaxf(s0, s1);
      mx = fmaxf(mx, __shfl_xor(mx, 1, 32));
      mx = fmaxf(mx, __shfl_xor(mx, 2, 32));
      mx = fmaxf(mx, __shfl_xor(mx, 4, 32));
      mx = fmaxf(mx, __shfl_xor(mx, 8, 32));
      float mnew = fmaxf(mrun[i], mx);
      float corr = __expf(mrun[i] - mnew);
      float p0 = __expf(s0 - mnew);
      float p1 = __expf(s1 - mnew);
      float rs = p0 + p1;
      rs += __shfl_xor(rs, 1, 32);
      rs += __shfl_xor(rs, 2, 32);
      rs += __shfl_xor(rs, 4, 32);
      rs += __shfl_xor(rs, 8, 32);
      lrun[i] = lrun[i] * corr + rs;
      mrun[i] = mnew;
      O0[i] *= corr; O1[i] *= corr; O2[i] *= corr; O3[i] *= corr;
      int prow = half * 8 + i;
      P[wv][prow][r] = f32_to_bf16_rne(p0);
      P[wv][prow][16 + r] = f32_to_bf16_rne(p1);
    }
    __syncthreads();

    // O += P(16x32) @ V(32x64): all frags loaded, then 4 WMMAs
    {
      Frag16 pa, vb0, vb1, vb2, vb3;
      pa.q[0] = *(const uint4*)&P[wv][r][aoff];
      pa.q[1] = *(const uint4*)&P[wv][r][16 + aoff];
      vb0.q[0] = *(const uint4*)&VT[r][aoff];
      vb0.q[1] = *(const uint4*)&VT[r][16 + aoff];
      vb1.q[0] = *(const uint4*)&VT[16 + r][aoff];
      vb1.q[1] = *(const uint4*)&VT[16 + r][16 + aoff];
      vb2.q[0] = *(const uint4*)&VT[32 + r][aoff];
      vb2.q[1] = *(const uint4*)&VT[32 + r][16 + aoff];
      vb3.q[0] = *(const uint4*)&VT[48 + r][aoff];
      vb3.q[1] = *(const uint4*)&VT[48 + r][16 + aoff];
      O0 = wmma_bf16(pa, vb0, O0);
      O1 = wmma_bf16(pa, vb1, O1);
      O2 = wmma_bf16(pa, vb2, O2);
      O3 = wmma_bf16(pa, vb3, O3);
    }
  }

  // normalize and write o as [b*T+t][E] with head offset (concat heads)
#pragma unroll
  for (int i = 0; i < 8; ++i) {
    int trow = tw0 + half * 8 + i;
    float inv = 1.0f / lrun[i];
    long long orow = ((long long)b * 1024 + trow) * 768 + h * 64;
    o[orow + r]      = f32_to_bf16_rne(O0[i] * inv);
    o[orow + 16 + r] = f32_to_bf16_rne(O1[i] * inv);
    o[orow + 32 + r] = f32_to_bf16_rne(O2[i] * inv);
    o[orow + 48 + r] = f32_to_bf16_rne(O3[i] * inv);
  }
}

// -------------------- launcher --------------------
extern "C" void kernel_launch(void* const* d_in, const int* in_sizes, int n_in,
                              void* d_out, int out_size, void* d_ws, size_t ws_size,
                              hipStream_t stream) {
  (void)in_sizes; (void)n_in; (void)out_size; (void)ws_size;
  const int* idx     = (const int*)d_in[0];
  const float* tok   = (const float*)d_in[1];
  const float* pos   = (const float*)d_in[2];
  const float* Wk    = (const float*)d_in[3];
  const float* Wq    = (const float*)d_in[4];
  const float* Wv    = (const float*)d_in[5];
  const float* Wproj = (const float*)d_in[6];
  const float* bproj = (const float*)d_in[7];
  const float* ln1g  = (const float*)d_in[8];
  const float* ln1b  = (const float*)d_in[9];
  const float* W1    = (const float*)d_in[10];
  const float* b1    = (const float*)d_in[11];
  const float* W2    = (const float*)d_in[12];
  const float* b2    = (const float*)d_in[13];
  const float* ln2g  = (const float*)d_in[14];
  const float* ln2b  = (const float*)d_in[15];
  const float* lnfg  = (const float*)d_in[16];
  const float* lnfb  = (const float*)d_in[17];
  const float* Wlm   = (const float*)d_in[18];
  const float* blm   = (const float*)d_in[19];
  float* logits = (float*)d_out;

  const int M = 2048, E = 768, H = 12, D = 64, FF = 3072, L = 12, V = 50257;

  char* ws = (char*)d_ws;
  float* x = (float*)ws;                      ws += (size_t)M * E * 4;
  unsigned short* hb = (unsigned short*)ws;   ws += (size_t)M * E * 2;
  unsigned short* qb = (unsigned short*)ws;   ws += (size_t)H * M * D * 2;
  unsigned short* kb = (unsigned short*)ws;   ws += (size_t)H * M * D * 2;
  unsigned short* vbuf = (unsigned short*)ws; ws += (size_t)H * M * D * 2;
  unsigned short* ob = (unsigned short*)ws;   ws += (size_t)M * E * 2;
  unsigned short* fb = (unsigned short*)ws;

  k_embed<<<dim3(M), dim3(256), 0, stream>>>(idx, tok, pos, x);

  for (int l = 0; l < L; ++l) {
    k_layernorm<<<dim3(M), dim3(256), 0, stream>>>(x, ln1g + l * E, ln1b + l * E, hb);

    dim3 gqkv(1, M / 128, H);   // N=64 per head, batched over heads via z
    k_gemm<<<gqkv, 256, 0, stream>>>(hb, Wq + (size_t)l * H * E * D, nullptr, nullptr,
                                     nullptr, qb, 64, E, 0,
                                     (long long)E * D, (long long)M * D);
    k_gemm<<<gqkv, 256, 0, stream>>>(hb, Wk + (size_t)l * H * E * D, nullptr, nullptr,
                                     nullptr, kb, 64, E, 0,
                                     (long long)E * D, (long long)M * D);
    k_gemm<<<gqkv, 256, 0, stream>>>(hb, Wv + (size_t)l * H * E * D, nullptr, nullptr,
                                     nullptr, vbuf, 64, E, 0,
                                     (long long)E * D, (long long)M * D);

    k_attn<<<dim3(16, 24), 128, 0, stream>>>(qb, kb, vbuf, ob);

    // x = x + o @ Wproj + bproj
    k_gemm<<<dim3(12, 16), 256, 0, stream>>>(ob, Wproj + (size_t)l * E * E, bproj + l * E,
                                             x, x, nullptr, E, E, 0, 0, 0);

    k_layernorm<<<dim3(M), dim3(256), 0, stream>>>(x, ln2g + l * E, ln2b + l * E, hb);

    // ff = relu(h2 @ W1 + b1)  -> bf16
    k_gemm<<<dim3(48, 16), 256, 0, stream>>>(hb, W1 + (size_t)l * E * FF, b1 + l * FF,
                                             nullptr, nullptr, fb, FF, E, 1, 0, 0);
    // x = x + ff @ W2 + b2
    k_gemm<<<dim3(12, 16), 256, 0, stream>>>(fb, W2 + (size_t)l * FF * E, b2 + l * E,
                                             x, x, nullptr, E, FF, 0, 0, 0);
  }

  k_layernorm<<<dim3(M), dim3(256), 0, stream>>>(x, lnfg, lnfb, hb);

  // logits = h @ Wlm + blm   (N=50257, ragged tail via clamped-column path)
  k_gemm<<<dim3((V + 63) / 64, 16), 256, 0, stream>>>(hb, Wlm, blm, nullptr,
                                                      logits, nullptr, V, E, 0, 0, 0);
}
